// MultiScalePointNetSetAbstraction_59176059404749
// MI455X (gfx1250) — matrix operand order repaired and checked by reading, not compiled, not run on hardware
//
#include <hip/hip_runtime.h>
#include <hip/hip_bf16.h>
#include <cstddef>

// ---------------------------------------------------------------------------
// MultiScale PointNet Set Abstraction for MI455X (gfx1250, wave32, WMMA)
//   B=8, N=4096, C=O=64, radii (0.1, 0.2), nsamples (32, 64)
// ---------------------------------------------------------------------------

#define BQ_B 8
#define BQ_N 4096
#define BQ_C 64
#define BQ_O 64
#define KPAD 96                 // 64 pts channels + 3 xyz + 29 zero pad
#define INV_SQRT_BN 0.9999950000374997f

// CDNA5 async global->LDS path (ASYNCcnt). Guarded: falls back to the
// sync gather if this toolchain does not declare the builtins.
#if defined(__AMDGCN__) && \
    __has_builtin(__builtin_amdgcn_global_load_async_to_lds_b64) && \
    __has_builtin(__builtin_amdgcn_s_wait_asynccnt)
#define BQ_ASYNC 1
#else
#define BQ_ASYNC 0
#endif

typedef __attribute__((ext_vector_type(16))) __bf16 v16bf;
typedef __attribute__((ext_vector_type(8)))  __bf16 v8bf;
typedef __attribute__((ext_vector_type(8)))  float  v8f;
typedef int v2i32 __attribute__((vector_size(8)));   // builtin's b64 payload type

__device__ __forceinline__ unsigned short f2bf(float f) {
    unsigned int u = __float_as_uint(f);
    u += 0x7FFFu + ((u >> 16) & 1u);          // round-to-nearest-even
    return (unsigned short)(u >> 16);
}

// ---------------------------------------------------------------------------
// Kernel 1: points (B,C,N) f32  ->  pts_t (B,N,C) bf16   (coalesced reads)
// ---------------------------------------------------------------------------
__global__ void transpose_to_bf16_kernel(const float* __restrict__ pts,
                                         unsigned short* __restrict__ out) {
    size_t i = (size_t)blockIdx.x * blockDim.x + threadIdx.x;   // b*C*N + c*N + n
    if (i >= (size_t)BQ_B * BQ_C * BQ_N) return;
    int n = (int)(i & (BQ_N - 1));
    size_t bc = i >> 12;
    int c = (int)(bc & (BQ_C - 1));
    int b = (int)(bc >> 6);
    out[(((size_t)b * BQ_N + n) << 6) + c] = f2bf(pts[i]);
}

// ---------------------------------------------------------------------------
// Kernel 2: fused dual-radius ball query. One wave32 per center point.
// Ordered scan over candidates j=0..N-1 in chunks of 32; ballot + prefix
// popcount compaction preserves ascending-index order (matches the
// where->sort->slice trick in the reference). Pads with the first hit.
// ---------------------------------------------------------------------------
__global__ void ball_query_kernel(const float* __restrict__ xyz,
                                  int* __restrict__ idx0,   // (B*N, 32) r=0.1
                                  int* __restrict__ idx1) { // (B*N, 64) r=0.2
    const float R0SQ = 0.1f * 0.1f;
    const float R1SQ = 0.2f * 0.2f;
    const int lane = threadIdx.x & 31;
    const int wave = threadIdx.x >> 5;
    const int g = blockIdx.x * 8 + wave;          // center id in [0, B*N)
    const int b = g >> 12;
    const int n = g & (BQ_N - 1);

    const float* base = xyz + (size_t)b * BQ_N * 3;
    const float cx = base[n * 3 + 0];
    const float cy = base[n * 3 + 1];
    const float cz = base[n * 3 + 2];

    int cnt0 = 0, cnt1 = 0, first0 = -1, first1 = -1;
    const unsigned below = (lane == 0) ? 0u : ((1u << lane) - 1u);

    for (int j0 = 0; j0 < BQ_N; j0 += 32) {
        const int j = j0 + lane;
        const float dx = base[j * 3 + 0] - cx;
        const float dy = base[j * 3 + 1] - cy;
        const float dz = base[j * 3 + 2] - cz;
        const float d = dx * dx + dy * dy + dz * dz;

        const unsigned m0 = (unsigned)__ballot(d <= R0SQ);
        const unsigned m1 = (unsigned)__ballot(d <= R1SQ);
        if (first1 < 0 && m1) first1 = j0 + (__ffs((int)m1) - 1);
        if (first0 < 0 && m0) first0 = j0 + (__ffs((int)m0) - 1);

        if (cnt0 < 32 && ((m0 >> lane) & 1u)) {
            const int p = cnt0 + __popc(m0 & below);
            if (p < 32) idx0[((size_t)g << 5) + p] = j;
        }
        if (cnt1 < 64 && ((m1 >> lane) & 1u)) {
            const int p = cnt1 + __popc(m1 & below);
            if (p < 64) idx1[((size_t)g << 6) + p] = j;
        }
        cnt0 += __popc(m0);
        cnt1 += __popc(m1);
        if (cnt0 >= 32 && cnt1 >= 64) break;      // wave-uniform
    }
    cnt0 = min(cnt0, 32);
    cnt1 = min(cnt1, 64);
    for (int p = cnt0 + lane; p < 32; p += 32) idx0[((size_t)g << 5) + p] = first0;
    for (int p = cnt1 + lane; p < 64; p += 32) idx1[((size_t)g << 6) + p] = first1;
}

// ---------------------------------------------------------------------------
// Kernel 3: grouped conv + neighbor-max + BN + ReLU, WMMA bf16 path.
//   Per wave / center:  feat (S x 96 bf16, LDS)  @  Wt (96 x 64 bf16, LDS)
//   tiled as 16x16x32 v_wmma_f32_16x16x32_bf16; running max over M-tiles.
//   Feature rows are gathered with GLOBAL_LOAD_ASYNC_TO_LDS_B64 (ASYNCcnt)
//   when available: no VGPR round-trip, overlaps with the WMMA pipeline.
// Block = 128 threads (4 waves); weight tile built once, reused 8 iters.
// ---------------------------------------------------------------------------
template <int S, bool FINAL>
__global__ void __launch_bounds__(128)
scale_wmma_kernel(const float* __restrict__ xyz,
                  const unsigned short* __restrict__ pts_bf16, // (B,N,64)
                  const int* __restrict__ idx,                 // (B*N, S)
                  const float* __restrict__ w,                 // (64, 67)
                  const float* __restrict__ bias,
                  const float* __restrict__ gamma,
                  const float* __restrict__ beta,
                  unsigned short* __restrict__ out_bf16,       // (B,N,64)
                  float* __restrict__ out_f32) {               // (B,64,N)
    constexpr int ITER = 8;
    // LDS: weights Bt[o=64][k=96] (transposed so a lane's 16 B elements are
    // two contiguous 16B reads), plus per-wave feat tiles [S][96].
    __shared__ __align__(32) unsigned short s_w[BQ_O * KPAD];
    __shared__ __align__(32) unsigned short s_feat[4 * S * KPAD];

    const int lane = threadIdx.x & 31;
    const int wave = threadIdx.x >> 5;
    const int row = lane & 15;
    const int half = lane >> 4;

    __builtin_prefetch(w, 0, 3);                 // global_prefetch_b8
    __builtin_prefetch(gamma, 0, 3);

    // Build padded/transposed weight tile: Bt[o][k]
    //   k < 64  -> w[o, 3+k]   (point-feature channels)
    //   64..66  -> w[o, k-64]  (xyz-delta channels)
    //   67..95  -> 0           (K padding)
    for (int i = threadIdx.x; i < BQ_O * KPAD; i += blockDim.x) {
        const int o = i / KPAD, k = i % KPAD;
        float wv = 0.0f;
        if (k < 64)      wv = w[o * 67 + 3 + k];
        else if (k < 67) wv = w[o * 67 + (k - 64)];
        s_w[o * KPAD + k] = f2bf(wv);
    }
    __syncthreads();

    unsigned short* feat = s_feat + wave * S * KPAD;

    for (int it = 0; it < ITER; ++it) {
        const int g = (blockIdx.x * ITER + it) * 4 + wave;    // center id
        const int b = g >> 12;
        const int n = g & (BQ_N - 1);

        const int idxA = idx[(size_t)g * S + lane];
        const int idxB = (S == 64) ? idx[(size_t)g * S + 32 + lane] : 0;

        const float cx = xyz[((size_t)b * BQ_N + n) * 3 + 0];
        const float cy = xyz[((size_t)b * BQ_N + n) * 3 + 1];
        const float cz = xyz[((size_t)b * BQ_N + n) * 3 + 2];

        // ---- stage point-feature channels (64 bf16 per neighbor row) ----
#if BQ_ASYNC
        // Two neighbors per step: lane half selects neighbor, row*4 selects
        // the 4-channel chunk. Async copy global -> LDS, tracked by ASYNCcnt.
        for (int s = 0; s < S; s += 2) {
            const int sl = s + half;
            const int j = (sl < 32) ? __shfl(idxA, sl, 32)
                                    : __shfl(idxB, sl - 32, 32);
            const unsigned short* srow =
                pts_bf16 + (((size_t)b * BQ_N + j) << 6) + row * 4;
            __builtin_amdgcn_global_load_async_to_lds_b64(
                (__attribute__((address_space(1))) v2i32*)srow,
                (__attribute__((address_space(3))) v2i32*)(feat + sl * KPAD + row * 4),
                0, 0);
        }
#else
        for (int s = 0; s < S; ++s) {
            const int j = (s < 32) ? __shfl(idxA, s, 32) : __shfl(idxB, s - 32, 32);
            const unsigned int* src =
                (const unsigned int*)(pts_bf16 + (((size_t)b * BQ_N + j) << 6));
            ((unsigned int*)(feat + s * KPAD))[lane] = src[lane];  // 2 ch/lane
        }
#endif
        // ---- xyz deltas (k=64..66) + zero K-padding (k=67..95) ----------
        for (int s = 0; s < S; ++s) {
            const int j = (s < 32) ? __shfl(idxA, s, 32) : __shfl(idxB, s - 32, 32);
            if (lane < 3) {
                const float cv = (lane == 0) ? cx : (lane == 1) ? cy : cz;
                const float dv = xyz[((size_t)b * BQ_N + j) * 3 + lane] - cv;
                feat[s * KPAD + 64 + lane] = f2bf(dv);
            } else {
                feat[s * KPAD + 64 + lane] = 0;                    // 67..95 pad
            }
        }
#if BQ_ASYNC
        __builtin_amdgcn_s_wait_asynccnt(0);      // s_wait_asynccnt 0
#endif

        // ---- WMMA: running max over sample rows -------------------------
        float mx[4] = {-3.0e38f, -3.0e38f, -3.0e38f, -3.0e38f};
#pragma unroll
        for (int mt = 0; mt < S / 16; ++mt) {
            const unsigned short* rp = feat + (mt * 16 + row) * KPAD;
            v16bf a[3];
#pragma unroll
            for (int k = 0; k < 3; ++k) {
                // A 16x32 bf16 layout: e<8 -> K=k*32+half*8+e ; e>=8 -> +16
                v8bf* ap = (v8bf*)&a[k];
                ap[0] = *(const v8bf*)(rp + k * 32 + half * 8);
                ap[1] = *(const v8bf*)(rp + k * 32 + 16 + half * 8);
            }
#pragma unroll
            for (int nt = 0; nt < 4; ++nt) {
                v8f acc = {};
#pragma unroll
                for (int k = 0; k < 3; ++k) {
                    // B 32x16 bf16: lane -> N=row, e<8 -> K=k*32+half*8+e ; e>=8 -> +16
                    v16bf bt;
                    v8bf* bp = (v8bf*)&bt;
                    bp[0] = *(const v8bf*)(s_w + (nt * 16 + row) * KPAD + k * 32 + half * 8);
                    bp[1] = *(const v8bf*)(s_w + (nt * 16 + row) * KPAD + k * 32 + 16 + half * 8);
                    acc = __builtin_amdgcn_wmma_f32_16x16x32_bf16(
                        false, a[k], false, bt, (short)0, acc, false, false);
                }
                // max over the 8 M-rows this lane-half holds
                float m = fmaxf(fmaxf(fmaxf(acc[0], acc[1]), fmaxf(acc[2], acc[3])),
                                fmaxf(fmaxf(acc[4], acc[5]), fmaxf(acc[6], acc[7])));
                mx[nt] = fmaxf(mx[nt], m);
            }
        }
        // combine lane halves (M 0..7 vs 8..15 per tile)
#pragma unroll
        for (int nt = 0; nt < 4; ++nt)
            mx[nt] = fmaxf(mx[nt], __shfl_xor(mx[nt], 16, 32));

        if (half == 0) {
#pragma unroll
            for (int nt = 0; nt < 4; ++nt) {
                const int o = nt * 16 + row;
                float v = (mx[nt] + bias[o]) * INV_SQRT_BN * gamma[o] + beta[o];
                v = fmaxf(v, 0.0f);
                if (FINAL) {
                    out_f32[(((size_t)b << 6) + o) * BQ_N + n] = v;
                } else {
                    out_bf16[(((size_t)b * BQ_N + n) << 6) + o] = f2bf(v);
                }
            }
        }
    }
}

// ---------------------------------------------------------------------------
// Host-side launch
// ---------------------------------------------------------------------------
extern "C" void kernel_launch(void* const* d_in, const int* in_sizes, int n_in,
                              void* d_out, int out_size, void* d_ws, size_t ws_size,
                              hipStream_t stream) {
    const float* xyz     = (const float*)d_in[0];   // (8,4096,3)
    const float* points  = (const float*)d_in[1];   // (8,64,4096)
    const float* conv0_w = (const float*)d_in[2];   // (64,67)
    const float* conv0_b = (const float*)d_in[3];
    const float* bn0_g   = (const float*)d_in[4];
    const float* bn0_b   = (const float*)d_in[5];
    const float* conv1_w = (const float*)d_in[6];   // (64,67)
    const float* conv1_b = (const float*)d_in[7];
    const float* bn1_g   = (const float*)d_in[8];
    const float* bn1_b   = (const float*)d_in[9];

    float* out_xyz = (float*)d_out;                         // 98304 floats
    float* out_h   = (float*)d_out + (size_t)BQ_B * BQ_N * 3;

    char* ws = (char*)d_ws;
    int*            idx0 = (int*)(ws);                        //  4 MiB
    int*            idx1 = (int*)(ws + ((size_t)4  << 20));   //  8 MiB
    unsigned short* pts0 = (unsigned short*)(ws + ((size_t)12 << 20)); // 4 MiB
    unsigned short* h0   = (unsigned short*)(ws + ((size_t)16 << 20)); // 4 MiB

    // Output 0: pass-through xyz.
    (void)hipMemcpyAsync(out_xyz, xyz, (size_t)BQ_B * BQ_N * 3 * sizeof(float),
                         hipMemcpyDeviceToDevice, stream);

    // points (B,C,N) f32 -> (B,N,C) bf16
    {
        const size_t total = (size_t)BQ_B * BQ_C * BQ_N;
        transpose_to_bf16_kernel<<<dim3((unsigned)((total + 255) / 256)), dim3(256),
                                   0, stream>>>(points, pts0);
    }

    // Dual-radius ball query: one wave per center, 8 waves/block.
    ball_query_kernel<<<dim3(BQ_B * BQ_N / 8), dim3(256), 0, stream>>>(
        xyz, idx0, idx1);

    // Pass 0: r=0.1, S=32 -> h0 (bf16).   32768 centers = 1024 blk * 4 waves * 8 it
    scale_wmma_kernel<32, false><<<dim3(1024), dim3(128), 0, stream>>>(
        xyz, pts0, idx0, conv0_w, conv0_b, bn0_g, bn0_b, h0, nullptr);

    // Pass 1: r=0.2, S=64 -> final output (f32, transposed to (B,64,N)).
    scale_wmma_kernel<64, true><<<dim3(1024), dim3(128), 0, stream>>>(
        xyz, h0, idx1, conv1_w, conv1_b, bn1_g, bn1_b, nullptr, out_h);
}
